// CausalSelfAttention_8443905704568
// MI455X (gfx1250) — matrix operand order repaired and checked by reading, not compiled, not run on hardware
//
#include <hip/hip_runtime.h>

#define T_SEQ 2048
#define C_EMB 4096
#define C3    12288
#define NH    32
#define HD    128
#define AT    10

typedef __attribute__((ext_vector_type(16))) __bf16       v16bf;
typedef __attribute__((ext_vector_type(8)))  float        v8f;
typedef __attribute__((ext_vector_type(4))) unsigned int  u32x4;
typedef __attribute__((ext_vector_type(2))) unsigned int  u32x2;
typedef __attribute__((ext_vector_type(4)))  float        f32x4;
typedef int v4i_gcc __attribute__((vector_size(16)));     // matches builtin param

struct Frag { unsigned int r[8]; };

__device__ inline unsigned short f2bf(float f) {
  unsigned u = __builtin_bit_cast(unsigned, f);
  u += 0x7fffu + ((u >> 16) & 1u);           // round-to-nearest-even
  return (unsigned short)(u >> 16);
}
__device__ inline float bf2f(unsigned short h) {
  unsigned u = ((unsigned)h) << 16;
  return __builtin_bit_cast(float, u);
}
__device__ inline float bflo(unsigned u) { return bf2f((unsigned short)(u & 0xffffu)); }
__device__ inline float bfhi(unsigned u) { return bf2f((unsigned short)(u >> 16)); }
__device__ inline unsigned pack2(float lo, float hi) {
  return (unsigned)f2bf(lo) | ((unsigned)f2bf(hi) << 16);
}
__device__ inline v8f zero8() {
  v8f z;
#pragma unroll
  for (int i = 0; i < 8; ++i) z[i] = 0.f;
  return z;
}
__device__ inline v8f wmma_bf16(const Frag& a, const Frag& b, v8f c) {
  v16bf av = __builtin_bit_cast(v16bf, a);
  v16bf bv = __builtin_bit_cast(v16bf, b);
  return __builtin_amdgcn_wmma_f32_16x16x32_bf16(false, av, false, bv,
                                                 (short)0, c, false, false);
}

// --- CDNA5 async global->LDS copy (ASYNCcnt path), guarded with fallback ----
#define HAS_ASYNC_LDS  __has_builtin(__builtin_amdgcn_global_load_async_to_lds_b128)
#define HAS_ASYNC_WAIT __has_builtin(__builtin_amdgcn_s_wait_asynccnt)

__device__ inline void copy16(unsigned int* lds_dst, const unsigned short* gsrc) {
#if HAS_ASYNC_LDS
  __builtin_amdgcn_global_load_async_to_lds_b128(
      (__attribute__((address_space(1))) v4i_gcc*)gsrc,
      (__attribute__((address_space(3))) v4i_gcc*)lds_dst, 0, 0);
#else
  *(u32x4*)lds_dst = *(const u32x4*)gsrc;
#endif
}
__device__ inline void copy_wait() {
#if HAS_ASYNC_LDS && HAS_ASYNC_WAIT
  __builtin_amdgcn_s_wait_asynccnt(0);
#endif
}

// ---------------------------------------------------------------------------
// fp32 -> packed bf16 conversion (one pass; halves all later weight traffic)
// ---------------------------------------------------------------------------
__global__ void __launch_bounds__(256) f32_to_bf16_kernel(
    const float* __restrict__ src, unsigned short* __restrict__ dst) {
  size_t i = ((size_t)blockIdx.x * 256 + threadIdx.x) * 8;
  f32x4 a = *(const f32x4*)(src + i);
  f32x4 b = *(const f32x4*)(src + i + 4);
  u32x4 p;
  p[0] = pack2(a[0], a[1]);
  p[1] = pack2(a[2], a[3]);
  p[2] = pack2(b[0], b[1]);
  p[3] = pack2(b[2], b[3]);
  *(u32x4*)(dst + i) = p;
}

// ---------------------------------------------------------------------------
// GEMM: out[m][n] = sum_k A[m][k] * Bw[n][k]   (A,Bw bf16; bf16 WMMA)
// Block tile 128x128, 8 waves, wave tile 64x32. Double-buffered LDS staging
// via async global->LDS b128 copies (row stride 20 dwords, 16B aligned).
// ---------------------------------------------------------------------------
template <typename OutT>
__global__ void __launch_bounds__(256) gemm_bf16_wmma(
    const unsigned short* __restrict__ A, const unsigned short* __restrict__ Bw,
    OutT* __restrict__ out, int M, int N, int K) {
  __shared__ unsigned int As[2][128 * 20];
  __shared__ unsigned int Bs[2][128 * 20];
  const int tid  = threadIdx.x;
  const int lane = tid & 31, wid = tid >> 5;
  const int l16  = lane & 15, hp = lane >> 4;
  const int waveM = wid >> 2, waveN = wid & 3;
  const int m0 = blockIdx.y * 128, n0 = blockIdx.x * 128;

  auto stage = [&](int buf, int k0) {
#pragma unroll
    for (int i = 0; i < 2; ++i) {
      int c = tid * 2 + i;           // 512 chunks of 16B per 128x32 tile
      int row = c >> 2, c4 = c & 3;
      copy16(&As[buf][row * 20 + c4 * 4],
             A + (size_t)(m0 + row) * K + k0 + c4 * 8);
      copy16(&Bs[buf][row * 20 + c4 * 4],
             Bw + (size_t)(n0 + row) * K + k0 + c4 * 8);
    }
  };

  v8f acc[4][2];
#pragma unroll
  for (int fm = 0; fm < 4; ++fm)
#pragma unroll
    for (int fn = 0; fn < 2; ++fn) acc[fm][fn] = zero8();

  stage(0, 0);
  copy_wait();
  __syncthreads();

  for (int k0 = 0; k0 < K; k0 += 32) {
    const int buf = (k0 >> 5) & 1;
    if (k0 + 32 < K) stage(buf ^ 1, k0 + 32);   // prefetch next tile (async)

    Frag a[4], b[2];
#pragma unroll
    for (int fm = 0; fm < 4; ++fm) {
      int row = waveM * 64 + fm * 16 + l16;
#pragma unroll
      for (int j = 0; j < 8; ++j)
        a[fm].r[j] = As[buf][row * 20 + ((j < 4) ? j : j + 4) + 4 * hp];
    }
#pragma unroll
    for (int fn = 0; fn < 2; ++fn) {
      int nrow = waveN * 32 + fn * 16 + l16;
#pragma unroll
      for (int j = 0; j < 8; ++j)
        b[fn].r[j] = Bs[buf][nrow * 20 + ((j < 4) ? j : j + 4) + 4 * hp];
    }
#pragma unroll
    for (int fm = 0; fm < 4; ++fm)
#pragma unroll
      for (int fn = 0; fn < 2; ++fn)
        acc[fm][fn] = wmma_bf16(a[fm], b[fn], acc[fm][fn]);

    copy_wait();        // prefetch landed
    __syncthreads();    // everyone done reading buf; next-tile LDS visible
  }

#pragma unroll
  for (int fm = 0; fm < 4; ++fm)
#pragma unroll
    for (int fn = 0; fn < 2; ++fn)
#pragma unroll
      for (int j = 0; j < 8; ++j) {
        int row = m0 + waveM * 64 + fm * 16 + j + 8 * hp;
        int col = n0 + waveN * 32 + fn * 16 + l16;
        float v = acc[fm][fn][j];
        if constexpr (sizeof(OutT) == 2)
          ((unsigned short*)out)[(size_t)row * N + col] = f2bf(v);
        else
          ((float*)out)[(size_t)row * N + col] = v;
      }
}

// ---------------------------------------------------------------------------
// RoPE on q and k halves of qkv (bf16 in-place).
// ---------------------------------------------------------------------------
__global__ void __launch_bounds__(256) rope_kernel(unsigned short* qkv) {
  int gid = blockIdx.x * 256 + threadIdx.x;      // 2*T*NH*64 threads
  int j = gid & 63;
  int h = (gid >> 6) & 31;
  int t = (gid >> 11) & 2047;
  int which = gid >> 22;                          // 0 = q, 1 = k
  unsigned short* row = qkv + (size_t)t * C3 + which * C_EMB + h * HD;
  float x0 = bf2f(row[j]), x1 = bf2f(row[j + 64]);
  float ang = (float)t * __expf((float)j * (-9.210340371976184f / 64.0f));
  float c = __cosf(ang), s = __sinf(ang);
  row[j]      = f2bf(x0 * c - x1 * s);
  row[j + 64] = f2bf(x1 * c + x0 * s);
}

// ---------------------------------------------------------------------------
// Adapter k/v projection: ak/av[h][j][d] = wte[j] . w_attn[o]   (tiny GEMM)
// ---------------------------------------------------------------------------
__global__ void __launch_bounds__(256) adapter_kv_kernel(
    const float* __restrict__ wte, const float* __restrict__ w_attn,
    unsigned short* __restrict__ ak, unsigned short* __restrict__ av) {
  int gid = blockIdx.x * 256 + threadIdx.x;       // 2*NH*AT*HD = 81920
  int d  = gid & 127;
  int j  = (gid >> 7) % AT;
  int h  = (gid / (128 * AT)) & 31;
  int kv = gid / (128 * AT * 32);
  int o  = C_EMB + kv * C_EMB + h * HD + d;
  const float* wr = w_attn + (size_t)o * C_EMB;
  const float* ar = wte + (size_t)j * C_EMB;
  float s = 0.f;
  for (int c = 0; c < C_EMB; c += 4) {
    f32x4 a = *(const f32x4*)(ar + c);
    f32x4 b = *(const f32x4*)(wr + c);
    s += a[0] * b[0] + a[1] * b[1] + a[2] * b[2] + a[3] * b[3];
  }
  unsigned short* dst = kv ? av : ak;
  dst[(h * AT + j) * HD + d] = f2bf(s);
}

// ---------------------------------------------------------------------------
// Flash-style causal attention: block = (128-query tile, head), 8 waves.
// Q/K fragments straight from global (K-pairs contiguous); V transposed in
// LDS; P staged per-wave in LDS (same-wave DS ordering). Output y in bf16.
// ---------------------------------------------------------------------------
__global__ void __launch_bounds__(256) attn_kernel(
    const unsigned short* __restrict__ qkv, unsigned short* __restrict__ y) {
  __shared__ unsigned int   Vt[128][65];       // [d][key-pair]
  __shared__ unsigned short Ps[8][16][130];    // [wave][row][key]
  const int qb = blockIdx.x, h = blockIdx.y;
  const int tid = threadIdx.x, lane = tid & 31, wid = tid >> 5;
  const int l16 = lane & 15, hp = lane >> 4;
  const int t0 = qb * 128;

  // Preload this wave's Q fragments (A-layout, 4 K-steps over hd=128).
  Frag qa[4];
  {
    const unsigned short* qrow =
        qkv + (size_t)(t0 + wid * 16 + l16) * C3 + h * HD;
#pragma unroll
    for (int ks = 0; ks < 4; ++ks) {
      u32x4 lo = *(const u32x4*)(qrow + ks * 32 + 8 * hp);
      u32x4 hi = *(const u32x4*)(qrow + ks * 32 + 16 + 8 * hp);
#pragma unroll
      for (int i = 0; i < 4; ++i) { qa[ks].r[i] = lo[i]; qa[ks].r[i + 4] = hi[i]; }
    }
  }

  v8f o[8];
  float m[8], lsum[8];
#pragma unroll
  for (int i = 0; i < 8; ++i) { o[i] = zero8(); m[i] = -1e30f; lsum[i] = 0.f; }

  for (int kb = 0; kb <= qb; ++kb) {
    const int k0 = kb * 128;
    // --- Stage V tile transposed: Vt[d][kk] = pack(V[2kk][d], V[2kk+1][d])
    {
      int kk = tid & 63, half = tid >> 6;
      int dbase = half * 32;
      const unsigned short* r0 =
          qkv + (size_t)(k0 + 2 * kk) * C3 + 2 * C_EMB + h * HD + dbase;
      const unsigned short* r1 = r0 + C3;
#pragma unroll
      for (int q4 = 0; q4 < 4; ++q4) {
        u32x4 a = *(const u32x4*)(r0 + q4 * 8);
        u32x4 b = *(const u32x4*)(r1 + q4 * 8);
#pragma unroll
        for (int i = 0; i < 4; ++i) {
          int d = dbase + q4 * 8 + i * 2;
          Vt[d][kk]     = (a[i] & 0xffffu) | (b[i] << 16);
          Vt[d + 1][kk] = (a[i] >> 16) | (b[i] & 0xffff0000u);
        }
      }
    }
    __syncthreads();

    // --- S = (Q K^T) * 1/sqrt(hd), K fragments from global
    v8f sfr[8];
#pragma unroll
    for (int fn = 0; fn < 8; ++fn) {
      v8f s = zero8();
      const unsigned short* krow =
          qkv + (size_t)(k0 + fn * 16 + l16) * C3 + C_EMB + h * HD;
#pragma unroll
      for (int ks = 0; ks < 4; ++ks) {
        Frag b;
        u32x4 lo = *(const u32x4*)(krow + ks * 32 + 8 * hp);
        u32x4 hi = *(const u32x4*)(krow + ks * 32 + 16 + 8 * hp);
#pragma unroll
        for (int i = 0; i < 4; ++i) { b.r[i] = lo[i]; b.r[i + 4] = hi[i]; }
        s = wmma_bf16(qa[ks], b, s);
      }
#pragma unroll
      for (int j = 0; j < 8; ++j) s[j] *= 0.088388347648318447f;
      if (kb == qb) {
        int keyl = fn * 16 + l16;
#pragma unroll
        for (int j = 0; j < 8; ++j) {
          int ql = wid * 16 + j + 8 * hp;
          if (keyl > ql) s[j] = -1e30f;
        }
      }
      sfr[fn] = s;
    }

    // --- Online softmax (rows are 16-lane groups; xor-shuffle reductions)
    float alpha[8];
#pragma unroll
    for (int j = 0; j < 8; ++j) {
      float rm = -1e30f;
#pragma unroll
      for (int fn = 0; fn < 8; ++fn) rm = fmaxf(rm, sfr[fn][j]);
      rm = fmaxf(rm, __shfl_xor(rm, 8));
      rm = fmaxf(rm, __shfl_xor(rm, 4));
      rm = fmaxf(rm, __shfl_xor(rm, 2));
      rm = fmaxf(rm, __shfl_xor(rm, 1));
      float mnew = fmaxf(m[j], rm);
      alpha[j] = __expf(m[j] - mnew);
      m[j] = mnew;
      float rs = 0.f;
#pragma unroll
      for (int fn = 0; fn < 8; ++fn) {
        float e = __expf(sfr[fn][j] - mnew);
        rs += e;
        Ps[wid][j + 8 * hp][fn * 16 + l16] = f2bf(e);   // stage P (bf16)
      }
      rs += __shfl_xor(rs, 8);
      rs += __shfl_xor(rs, 4);
      rs += __shfl_xor(rs, 2);
      rs += __shfl_xor(rs, 1);
      lsum[j] = lsum[j] * alpha[j] + rs;
    }
#pragma unroll
    for (int fo = 0; fo < 8; ++fo)
#pragma unroll
      for (int j = 0; j < 8; ++j) o[fo][j] *= alpha[j];

    // --- O += P @ V  (P from per-wave LDS staging, V from transposed LDS)
    const unsigned* prow = (const unsigned*)&Ps[wid][l16][0];
#pragma unroll
    for (int ks = 0; ks < 4; ++ks) {
      Frag pa;
#pragma unroll
      for (int j = 0; j < 8; ++j)
        pa.r[j] = prow[ks * 16 + ((j < 4) ? j : j + 4) + 4 * hp];
#pragma unroll
      for (int fo = 0; fo < 8; ++fo) {
        Frag vb;
#pragma unroll
        for (int j = 0; j < 8; ++j)
          vb.r[j] = Vt[fo * 16 + l16][ks * 16 + ((j < 4) ? j : j + 4) + 4 * hp];
        o[fo] = wmma_bf16(pa, vb, o[fo]);
      }
    }
    __syncthreads();   // protect Vt before next block's staging
  }

  // --- Normalize and write y (bf16)
#pragma unroll
  for (int fo = 0; fo < 8; ++fo)
#pragma unroll
    for (int j = 0; j < 8; ++j) {
      int row = t0 + wid * 16 + j + 8 * hp;
      int col = h * HD + fo * 16 + l16;
      y[(size_t)row * C_EMB + col] = f2bf(o[fo][j] / lsum[j]);
    }
}

// ---------------------------------------------------------------------------
// Adapter cross-attention: one wave per (t, h); aT=10 keys; adds into bf16 y.
// ---------------------------------------------------------------------------
__global__ void __launch_bounds__(256) adapter_attn_kernel(
    const unsigned short* __restrict__ qkv,
    const unsigned short* __restrict__ ak,
    const unsigned short* __restrict__ av,
    const float* __restrict__ gating, unsigned short* __restrict__ y) {
  int gw = blockIdx.x * 8 + (threadIdx.x >> 5);
  int lane = threadIdx.x & 31;
  int h = gw & 31;
  int t = gw >> 5;

  u32x2 qp = *(const u32x2*)(qkv + (size_t)t * C3 + h * HD + lane * 4);
  float qf[4] = { bflo(qp[0]), bfhi(qp[0]), bflo(qp[1]), bfhi(qp[1]) };

  float sc[AT];
#pragma unroll
  for (int j = 0; j < AT; ++j) {
    u32x2 kp = *(const u32x2*)(ak + (h * AT + j) * HD + lane * 4);
    float p = qf[0] * bflo(kp[0]) + qf[1] * bfhi(kp[0]) +
              qf[2] * bflo(kp[1]) + qf[3] * bfhi(kp[1]);
    p += __shfl_xor(p, 16);
    p += __shfl_xor(p, 8);
    p += __shfl_xor(p, 4);
    p += __shfl_xor(p, 2);
    p += __shfl_xor(p, 1);
    sc[j] = p * (1.0f / 64.0f);          // 1/sqrt(C) = 1/64
  }
  float mx = sc[0];
#pragma unroll
  for (int j = 1; j < AT; ++j) mx = fmaxf(mx, sc[j]);
  float den = 0.f, pj[AT];
#pragma unroll
  for (int j = 0; j < AT; ++j) { pj[j] = __expf(sc[j] - mx); den += pj[j]; }
  float g = gating[0] / den;

  float acc[4] = {0.f, 0.f, 0.f, 0.f};
#pragma unroll
  for (int j = 0; j < AT; ++j) {
    u32x2 vp = *(const u32x2*)(av + (h * AT + j) * HD + lane * 4);
    float w = g * pj[j];
    acc[0] += w * bflo(vp[0]);
    acc[1] += w * bfhi(vp[0]);
    acc[2] += w * bflo(vp[1]);
    acc[3] += w * bfhi(vp[1]);
  }
  unsigned short* yp = y + (size_t)t * C_EMB + h * HD + lane * 4;
  u32x2 cur = *(u32x2*)yp;
  u32x2 nv;
  nv[0] = pack2(bflo(cur[0]) + acc[0], bfhi(cur[0]) + acc[1]);
  nv[1] = pack2(bflo(cur[1]) + acc[2], bfhi(cur[1]) + acc[3]);
  *(u32x2*)yp = nv;
}

// ---------------------------------------------------------------------------
extern "C" void kernel_launch(void* const* d_in, const int* in_sizes, int n_in,
                              void* d_out, int out_size, void* d_ws,
                              size_t ws_size, hipStream_t stream) {
  const float* x      = (const float*)d_in[0];
  const float* w_attn = (const float*)d_in[1];
  const float* w_proj = (const float*)d_in[2];
  const float* wte    = (const float*)d_in[3];
  const float* gating = (const float*)d_in[4];
  float* out = (float*)d_out;

  char* ws = (char*)d_ws;
  size_t off = 0;
  unsigned short* qkv  = (unsigned short*)(ws + off); off += (size_t)T_SEQ * C3 * 2;     // 48 MB
  unsigned short* y    = (unsigned short*)(ws + off); off += (size_t)T_SEQ * C_EMB * 2;  // 16 MB
  unsigned short* xbf  = (unsigned short*)(ws + off); off += (size_t)T_SEQ * C_EMB * 2;  // 16 MB
  unsigned short* wabf = (unsigned short*)(ws + off); off += (size_t)C3 * C_EMB * 2;     // 96 MB
  unsigned short* wpbf = (unsigned short*)(ws + off); off += (size_t)C_EMB * C_EMB * 2;  // 32 MB
  unsigned short* ak   = (unsigned short*)(ws + off); off += (size_t)NH * AT * HD * 2;
  unsigned short* av   = (unsigned short*)(ws + off);

  // One-time fp32 -> bf16 conversion (saves ~1.5 GB of weight re-reads).
  f32_to_bf16_kernel<<<(T_SEQ * C_EMB) / 2048, 256, 0, stream>>>(x, xbf);
  f32_to_bf16_kernel<<<(C3 * C_EMB) / 2048, 256, 0, stream>>>(w_attn, wabf);
  f32_to_bf16_kernel<<<(C_EMB * C_EMB) / 2048, 256, 0, stream>>>(w_proj, wpbf);

  gemm_bf16_wmma<unsigned short>
      <<<dim3(C3 / 128, T_SEQ / 128), 256, 0, stream>>>(
          xbf, wabf, qkv, T_SEQ, C3, C_EMB);
  rope_kernel<<<(2 * T_SEQ * NH * 64) / 256, 256, 0, stream>>>(qkv);
  adapter_kv_kernel<<<(2 * NH * AT * HD) / 256, 256, 0, stream>>>(
      wte, w_attn, ak, av);
  attn_kernel<<<dim3(T_SEQ / 128, NH), 256, 0, stream>>>(qkv, y);
  adapter_attn_kernel<<<(T_SEQ * NH) / 8, 256, 0, stream>>>(
      qkv, ak, av, gating, y);
  gemm_bf16_wmma<float>
      <<<dim3(C_EMB / 128, T_SEQ / 128), 256, 0, stream>>>(
          y, wpbf, out, T_SEQ, C_EMB, C_EMB);
}
